// DenseGATLayer_33174327394798
// MI455X (gfx1250) — matrix operand order repaired and checked by reading, not compiled, not run on hardware
//
#include <hip/hip_runtime.h>

// ---------------------------------------------------------------------------
// DenseGAT layer for MI455X (gfx1250): bf16 WMMA everywhere, fp32 epilogues.
//   (0) fp32 -> bf16 conversions; weights stored TRANSPOSED (N-major) so all
//       WMMA B-fragments are contiguous 32B-per-lane loads.
//       adjacency mask bit-packed to u32 words (node_mask folded in).
//   (1) QKV GEMM -> Q,K row-major [B,H,N,128]; V transposed [B,H,128,N]
//   (2) flash attention, LDS-staged K/V tiles (async-to-LDS), bitmask scores
//   (3) out proj + bias + residual + LayerNorm + SiLU + node mask (fp32 out)
// ---------------------------------------------------------------------------

#define Bdim 4
#define Ndim 2048
#define Fdim 128
#define Hdim 8
#define Ddim 128
#define HD   1024
#define BN   (Bdim * Ndim)
#define EPSV 1e-5f
#define NEGF (-1e30f)
#define SCALE 0.08838834764831845f  // 1/sqrt(128)

typedef unsigned short us;
typedef __attribute__((ext_vector_type(16))) __bf16 v16bf;
typedef __attribute__((ext_vector_type(8)))  float  v8f;
typedef int v4i __attribute__((vector_size(16)));

union V16BF { v16bf v; us u[16]; };
union V8F   { v8f  v; float f[8]; };

#if defined(__has_builtin)
#if __has_builtin(__builtin_amdgcn_global_load_async_to_lds_b128)
#define HAVE_ASYNC_LDS 1
#endif
#endif

__device__ __forceinline__ us f2bf(float x) {
  unsigned int u = __float_as_uint(x);
  u += 0x7FFFu + ((u >> 16) & 1u);   // round-to-nearest-even
  return (us)(u >> 16);
}

// A fragment: 16x32 (MxK) bf16 from row-major src, leading dim ld (elements).
// lane m (0..15): K = 0..15; lane m+16: K = 16..31 (contiguous halves).
__device__ __forceinline__ v16bf load_a_frag(const us* src, int ld, int row0, int k0) {
  const int lane = threadIdx.x & 31;
  const int hf = lane >> 4;
  const int m  = lane & 15;
  const us* p0 = src + (size_t)(row0 + m) * ld + k0 + hf * 16;
  const us* p1 = p0 + 8;
  V16BF a;
#pragma unroll
  for (int e = 0; e < 8; ++e) a.u[e] = p0[e];
#pragma unroll
  for (int e = 0; e < 8; ++e) a.u[8 + e] = p1[e];
  return a.v;
}

// B fragment (32x16 KxN) from an N-major source S [ncols x K] row-major:
// element(k, n) = S[col0+n][k0+k].  Contraction is unit-stride: 32B per lane.
// lane n (0..15): K = 0..15; lane n+16: K = 16..31 (contiguous halves).
__device__ __forceinline__ v16bf load_bt_frag(const us* src, int ld, int col0, int k0) {
  const int lane = threadIdx.x & 31;
  const int hf = lane >> 4;
  const int n  = lane & 15;
  const us* p = src + (size_t)(col0 + n) * ld + k0 + hf * 16;
  V16BF bm;
#pragma unroll
  for (int e = 0; e < 16; ++e) bm.u[e] = p[e];
  return bm.v;
}

__device__ __forceinline__ v8f wmma_bf16(v16bf a, v16bf b, v8f c) {
  return __builtin_amdgcn_wmma_f32_16x16x32_bf16(false, a, false, b, (short)0, c,
                                                 false, false);
}

// ---------------------------------------------------------------------------
// (0a) straight fp32 -> bf16
// ---------------------------------------------------------------------------
__global__ __launch_bounds__(256) void cvt_bf16_kernel(const float* __restrict__ src,
                                                       us* __restrict__ dst, int n) {
  int i = blockIdx.x * 256 + threadIdx.x;
  if (i < n) dst[i] = f2bf(src[i]);
}

// (0b) fp32 [rows x cols] -> bf16 transposed [cols x rows]
__global__ __launch_bounds__(256) void cvt_bf16_t_kernel(const float* __restrict__ src,
                                                         us* __restrict__ dst,
                                                         int rows, int cols) {
  int i = blockIdx.x * 256 + threadIdx.x;
  if (i < rows * cols) {
    int r = i / cols, c = i - r * cols;
    dst[(size_t)c * rows + r] = f2bf(src[i]);
  }
}

// (0c) bit-pack adjacency: adjp[row][w] bit j = adj[row][32w+j]; node_mask of
// the query row folded in (invalid query -> all-zero row -> attn row zeroed).
__global__ __launch_bounds__(256) void pack_adj_kernel(const unsigned char* __restrict__ adj,
                                                       const unsigned char* __restrict__ nodem,
                                                       unsigned* __restrict__ adjp) {
  int i = blockIdx.x * 256 + threadIdx.x;   // one u32 word per thread
  if (i >= BN * 64) return;
  int row = i >> 6;                          // b*N + q
  const unsigned char* src = adj + (size_t)row * Ndim + (i & 63) * 32;
  unsigned m = 0;
#pragma unroll
  for (int j = 0; j < 32; ++j) m |= (src[j] ? 1u : 0u) << j;
  if (!nodem[row]) m = 0;
  adjp[i] = m;
}

// ---------------------------------------------------------------------------
// (1) QKV projection GEMM.  grid = (BN/16, 24), block = 256 (8 waves).
// WbT: 3 x [1024 x 128] (output-column-major weights).
// Q,K written row-major [B,H,N,128]; V written transposed [B,H,128,N].
// ---------------------------------------------------------------------------
__global__ __launch_bounds__(256) void qkv_kernel(const us* __restrict__ xb,
                                                  const us* __restrict__ WbT,
                                                  us* __restrict__ Qb,
                                                  us* __restrict__ Kb,
                                                  us* __restrict__ Vt) {
  const int wid = threadIdx.x >> 5;
  const int lane = threadIdx.x & 31;
  const int hf = lane >> 4, nl = lane & 15;
  const int m0 = blockIdx.x * 16;
  const int ct = blockIdx.y * 8 + wid;   // 0..191 column tiles of 3072
  const int w  = ct >> 6;                // 0=Q 1=K 2=V
  const int c  = (ct & 63) * 16;         // column within 1024
  const int h  = c >> 7;
  const int d0 = c & 127;
  const us* WT = WbT + (size_t)w * HD * Fdim;

  V8F acc;
#pragma unroll
  for (int j = 0; j < 8; ++j) acc.f[j] = 0.0f;
#pragma unroll
  for (int kk = 0; kk < Fdim; kk += 32) {
    v16bf a  = load_a_frag(xb, Fdim, m0, kk);
    v16bf bm = load_bt_frag(WT, Fdim, c, kk);
    acc.v = wmma_bf16(a, bm, acc.v);
  }

  if (w == 2) {
#pragma unroll
    for (int r = 0; r < 8; ++r) {
      int row = m0 + r + 8 * hf;
      int bb = row >> 11, n = row & (Ndim - 1);
      Vt[(((size_t)bb * Hdim + h) * Ddim + d0 + nl) * Ndim + n] = f2bf(acc.f[r]);
    }
  } else {
    us* dst = (w == 0) ? Qb : Kb;
#pragma unroll
    for (int r = 0; r < 8; ++r) {
      int row = m0 + r + 8 * hf;
      int bb = row >> 11, n = row & (Ndim - 1);
      dst[(((size_t)bb * Hdim + h) * Ndim + n) * Ddim + d0 + nl] = f2bf(acc.f[r]);
    }
  }
}

// ---------------------------------------------------------------------------
// LDS staging helper: copy 16B chunks global->LDS, async when available.
// ---------------------------------------------------------------------------
__device__ __forceinline__ void stage16(us* lds_dst, const us* gsrc) {
#if HAVE_ASYNC_LDS
  __builtin_amdgcn_global_load_async_to_lds_b128((v4i*)gsrc, (v4i*)lds_dst, 0, 0);
#else
  *(uint4*)lds_dst = *(const uint4*)gsrc;
#endif
}

__device__ __forceinline__ void stage_wait() {
#if HAVE_ASYNC_LDS
#if __has_builtin(__builtin_amdgcn_s_wait_asynccnt)
  __builtin_amdgcn_s_wait_asynccnt(0);
#else
  asm volatile("s_wait_asynccnt 0" ::: "memory");
#endif
#endif
}

// ---------------------------------------------------------------------------
// (2) Flash attention.  grid = (N/128, H, B), block = 256 (8 waves).
// K tile staged [32 keys x 128d]; V tile staged [128d x 32 keys] (from Vt).
// Mask read as one u32 bit-word per query row per 32-key block.
// ---------------------------------------------------------------------------
__global__ __launch_bounds__(256) void attn_kernel(const us* __restrict__ Qb,
                                                   const us* __restrict__ Kb,
                                                   const us* __restrict__ Vt,
                                                   const unsigned* __restrict__ adjp,
                                                   us* __restrict__ Ob) {
  __shared__ us Ksh[32 * 128];
  __shared__ us Vsh[128 * 32];
  __shared__ us Psh[8][16 * 32];

  const int b = blockIdx.z, h = blockIdx.y;
  const int wid = threadIdx.x >> 5;
  const int lane = threadIdx.x & 31;
  const int hf = lane >> 4, nl = lane & 15;
  const int q0 = blockIdx.x * 128 + wid * 16;

  const us* Qh = Qb + ((size_t)(b * Hdim + h)) * Ndim * Ddim;
  const us* Kh = Kb + ((size_t)(b * Hdim + h)) * Ndim * Ddim;
  const us* Vh = Vt + ((size_t)(b * Hdim + h)) * Ddim * Ndim;  // [128 x N]

  // Hoist the Q fragments for the whole key loop.
  v16bf aq[4];
#pragma unroll
  for (int kk = 0; kk < 4; ++kk) aq[kk] = load_a_frag(Qh, Ddim, q0, kk * 32);

  int qrow[8];
  size_t mrowbase[8];
#pragma unroll
  for (int r = 0; r < 8; ++r) {
    qrow[r] = q0 + r + 8 * hf;
    mrowbase[r] = ((size_t)b * Ndim + qrow[r]) * 64;   // adjp row base (words)
  }

  float mrow[8], lrow[8];
  V8F o[8];
#pragma unroll
  for (int r = 0; r < 8; ++r) { mrow[r] = NEGF; lrow[r] = 0.0f; }
#pragma unroll
  for (int t = 0; t < 8; ++t)
#pragma unroll
    for (int j = 0; j < 8; ++j) o[t].f[j] = 0.0f;

  for (int k0 = 0; k0 < Ndim; k0 += 32) {
    __syncthreads();
    // Stage K [32 x 128] and V [128 x 32] bf16 tiles into LDS.
    for (int i = threadIdx.x; i < 512; i += 256) {
      int rr = i >> 4, cc = (i & 15) * 8;           // K: 32 rows x 16 chunks
      stage16(&Ksh[rr * 128 + cc], &Kh[(size_t)(k0 + rr) * 128 + cc]);
      int dd = i >> 2, c2 = (i & 3) * 8;            // V: 128 rows x 4 chunks
      stage16(&Vsh[dd * 32 + c2], &Vh[(size_t)dd * Ndim + k0 + c2]);
    }
    stage_wait();
    __syncthreads();

    // S = Q @ K^T for 32 keys (two 16-key N-tiles).
    V8F s0, s1;
#pragma unroll
    for (int j = 0; j < 8; ++j) { s0.f[j] = 0.0f; s1.f[j] = 0.0f; }
#pragma unroll
    for (int kk = 0; kk < 4; ++kk) {
      v16bf bk0 = load_bt_frag(Ksh, 128, 0, kk * 32);
      v16bf bk1 = load_bt_frag(Ksh, 128, 16, kk * 32);
      s0.v = wmma_bf16(aq[kk], bk0, s0.v);
      s1.v = wmma_bf16(aq[kk], bk1, s1.v);
    }

    // Online softmax update (per query row r; lanes of a half share the row).
    const int widx = k0 >> 5;
    us* st = Psh[wid];
#pragma unroll
    for (int r = 0; r < 8; ++r) {
      unsigned wbits = adjp[mrowbase[r] + widx];
      float x0 = s0.f[r] * SCALE;
      float x1 = s1.f[r] * SCALE;
      x0 = ((wbits >> nl) & 1u) ? x0 : NEGF;
      x1 = ((wbits >> (nl + 16)) & 1u) ? x1 : NEGF;
      float t = fmaxf(x0, x1);
#pragma unroll
      for (int off = 8; off >= 1; off >>= 1) t = fmaxf(t, __shfl_xor(t, off, 32));
      float mn = fmaxf(mrow[r], t);
      float alpha = __expf(mrow[r] - mn);
      float p0 = __expf(x0 - mn);
      float p1 = __expf(x1 - mn);
      float rs = p0 + p1;
#pragma unroll
      for (int off = 8; off >= 1; off >>= 1) rs += __shfl_xor(rs, off, 32);
      lrow[r] = lrow[r] * alpha + rs;
      mrow[r] = mn;
#pragma unroll
      for (int t8 = 0; t8 < 8; ++t8) o[t8].f[r] *= alpha;
      int mloc = r + 8 * hf;
      st[mloc * 32 + nl] = f2bf(p0);
      st[mloc * 32 + 16 + nl] = f2bf(p1);
    }

    // O += P @ V   (P: 16x32 A-fragment from LDS; V: contiguous B^T frags)
    v16bf ap = load_a_frag(st, 32, 0, 0);
#pragma unroll
    for (int dt = 0; dt < 8; ++dt) {
      v16bf bv = load_bt_frag(Vsh, 32, dt * 16, 0);
      o[dt].v = wmma_bf16(ap, bv, o[dt].v);
    }
  }

  // Normalize; rows with no valid key (m stayed ~ -1e30) -> zeros.
  float inv[8];
#pragma unroll
  for (int r = 0; r < 8; ++r)
    inv[r] = (mrow[r] > -1e29f && lrow[r] > 0.0f) ? (1.0f / lrow[r]) : 0.0f;

#pragma unroll
  for (int dt = 0; dt < 8; ++dt) {
#pragma unroll
    for (int r = 0; r < 8; ++r) {
      float val = o[dt].f[r] * inv[r];
      Ob[((size_t)b * Ndim + qrow[r]) * HD + h * Ddim + dt * 16 + nl] = f2bf(val);
    }
  }
}

// ---------------------------------------------------------------------------
// (3) Output projection + bias + residual + LayerNorm + SiLU + node mask.
// WoT: [128 x 1024] (output-column-major).  grid = BN/16, block = 256.
// ---------------------------------------------------------------------------
__global__ __launch_bounds__(256) void outproj_kernel(const us* __restrict__ Ob,
                                                      const us* __restrict__ WoT,
                                                      const float* __restrict__ bo,
                                                      const float* __restrict__ x,
                                                      const float* __restrict__ gamma,
                                                      const float* __restrict__ beta,
                                                      const unsigned char* __restrict__ nodem,
                                                      float* __restrict__ out) {
  __shared__ float tile[16 * 128];
  __shared__ float mu_s[16], rs_s[16];

  const int wid = threadIdx.x >> 5;
  const int lane = threadIdx.x & 31;
  const int hf = lane >> 4, nl = lane & 15;
  const int m0 = blockIdx.x * 16;
  const int d0 = wid * 16;

  V8F acc;
#pragma unroll
  for (int j = 0; j < 8; ++j) acc.f[j] = 0.0f;
  for (int kk = 0; kk < HD; kk += 32) {
    v16bf a  = load_a_frag(Ob, HD, m0, kk);
    v16bf bm = load_bt_frag(WoT, HD, d0, kk);
    acc.v = wmma_bf16(a, bm, acc.v);
  }

#pragma unroll
  for (int r = 0; r < 8; ++r) {
    int lr = r + 8 * hf;
    int row = m0 + lr;
    int d = d0 + nl;
    tile[lr * 128 + d] = acc.f[r] + bo[d] + x[(size_t)row * Ddim + d];
  }
  __syncthreads();

  if (threadIdx.x < 16) {
    const float* rowp = &tile[threadIdx.x * 128];
    float s = 0.0f;
    for (int c = 0; c < 128; ++c) s += rowp[c];
    float mu = s * (1.0f / 128.0f);
    float v = 0.0f;
    for (int c = 0; c < 128; ++c) { float dd = rowp[c] - mu; v += dd * dd; }
    mu_s[threadIdx.x] = mu;
    rs_s[threadIdx.x] = rsqrtf(v * (1.0f / 128.0f) + EPSV);
  }
  __syncthreads();

#pragma unroll
  for (int i = 0; i < 8; ++i) {
    int e = threadIdx.x * 8 + i;
    int lr = e >> 7, c = e & 127;
    int row = m0 + lr;
    float v = (tile[e] - mu_s[lr]) * rs_s[lr] * gamma[c] + beta[c];
    float sg = 1.0f / (1.0f + __expf(-v));
    v *= sg;                                   // SiLU
    v *= nodem[row] ? 1.0f : 0.0f;             // node mask
    out[(size_t)row * Ddim + c] = v;
  }
}

// ---------------------------------------------------------------------------
extern "C" void kernel_launch(void* const* d_in, const int* in_sizes, int n_in,
                              void* d_out, int out_size, void* d_ws, size_t ws_size,
                              hipStream_t stream) {
  const float* x = (const float*)d_in[0];
  const unsigned char* adj = (const unsigned char*)d_in[1];
  const unsigned char* nodem = (const unsigned char*)d_in[2];
  const float* Wq = (const float*)d_in[3];
  const float* Wk = (const float*)d_in[4];
  const float* Wv = (const float*)d_in[5];
  const float* Wo = (const float*)d_in[6];
  const float* bo = (const float*)d_in[7];
  const float* gamma = (const float*)d_in[8];
  const float* beta = (const float*)d_in[9];
  (void)in_sizes; (void)n_in; (void)out_size; (void)ws_size;

  char* p = (char*)d_ws;
  us* xb  = (us*)p; p += (size_t)BN * Fdim * 2;             //  2 MB
  us* WbT = (us*)p; p += (size_t)3 * Fdim * HD * 2;         //  0.75 MB (transposed)
  us* WoT = (us*)p; p += (size_t)HD * Ddim * 2;             //  0.25 MB (transposed)
  unsigned* adjp = (unsigned*)p; p += (size_t)BN * 64 * 4;  //  2 MB (bit mask)
  us* Qb  = (us*)p; p += (size_t)Bdim * Hdim * Ndim * Ddim * 2;  // 16 MB
  us* Kb  = (us*)p; p += (size_t)Bdim * Hdim * Ndim * Ddim * 2;  // 16 MB
  us* Vt  = (us*)p; p += (size_t)Bdim * Hdim * Ddim * Ndim * 2;  // 16 MB (transposed)
  us* Ob  = (us*)p; p += (size_t)BN * HD * 2;                    // 16 MB

  const int nx = BN * Fdim;        // 1048576
  const int nw = Fdim * HD;        // 131072
  cvt_bf16_kernel<<<(nx + 255) / 256, 256, 0, stream>>>(x, xb, nx);
  cvt_bf16_t_kernel<<<(nw + 255) / 256, 256, 0, stream>>>(Wq, WbT, Fdim, HD);
  cvt_bf16_t_kernel<<<(nw + 255) / 256, 256, 0, stream>>>(Wk, WbT + nw, Fdim, HD);
  cvt_bf16_t_kernel<<<(nw + 255) / 256, 256, 0, stream>>>(Wv, WbT + 2 * nw, Fdim, HD);
  cvt_bf16_t_kernel<<<(nw + 255) / 256, 256, 0, stream>>>(Wo, WoT, HD, Ddim);
  pack_adj_kernel<<<(BN * 64 + 255) / 256, 256, 0, stream>>>(adj, nodem, adjp);

  qkv_kernel<<<dim3(BN / 16, 24), 256, 0, stream>>>(xb, WbT, Qb, Kb, Vt);
  attn_kernel<<<dim3(Ndim / 128, Hdim, Bdim), 256, 0, stream>>>(Qb, Kb, Vt, adjp, Ob);
  outproj_kernel<<<BN / 16, 256, 0, stream>>>(Ob, WoT, bo, x, gamma, beta, nodem,
                                              (float*)d_out);
}
